// sDCTFTParametrization_42434276884509
// MI455X (gfx1250) — compile-verified, matrix operand order checked by reading, not run on hardware
//
#include <hip/hip_runtime.h>
#include <hip/hip_bf16.h>
#include <math.h>

typedef __attribute__((ext_vector_type(8)))  _Float16 v8h;
typedef __attribute__((ext_vector_type(16))) _Float16 v16h;
typedef __attribute__((ext_vector_type(8)))  float    v8f;

#define MDIM  4096
#define NDIM  4096
#define NCOEF 768
#define ALPHA 16.0f

// ---------------------------------------------------------------------------
// Prep: for each (r, c) in [4096, 768] compute
//   Amat[r*768+c] = D_M[k_c, r] * dW[c] * ALPHA   (f16)
//   Bt  [r*768+c] = D_N[l_c, r]                   (f16)
// where flat index fi = k_c*4096 + l_c, and
//   D[k, t] = s(k) * cos(pi*(t+0.5)*k/4096), s(0)=1/64, s(k>0)=sqrt(2)/64.
// Exact integer range reduction: cos(pi*(2r+1)*k/8192) has period 16384 in
// the integer (2r+1)*k, so reduce with & 16383 before float conversion.
// ---------------------------------------------------------------------------
__global__ __launch_bounds__(256)
void prep_kernel(const float* __restrict__ dW, const int* __restrict__ fidx,
                 _Float16* __restrict__ Amat, _Float16* __restrict__ Bt) {
    const int idx = blockIdx.x * blockDim.x + threadIdx.x;   // 0 .. 4096*768-1
    const int c = idx % NCOEF;
    const int r = idx / NCOEF;

    const int fi = fidx[c];
    const int k  = fi >> 12;
    const int l  = fi & 4095;
    const float v = dW[c];

    const float S0     = 0.015625f;            // sqrt(1/4096)
    const float S1     = 0.02209708691f;       // sqrt(2/4096)
    const float PIDIV  = 3.834951969714103e-4f; // pi / 8192

    const int t2 = 2 * r + 1;                  // max 8191; products < 2^31
    const int ra = (t2 * k) & 16383;
    const int rb = (t2 * l) & 16383;

    const float a = (k == 0 ? S0 : S1) * cosf((float)ra * PIDIV) * v * ALPHA;
    const float b = (l == 0 ? S0 : S1) * cosf((float)rb * PIDIV);

    Amat[idx] = (_Float16)a;
    Bt[idx]   = (_Float16)b;
}

// ---------------------------------------------------------------------------
// GEMM: out[i,j] = weight[i,j] + sum_c Amat[i,c] * Bt[j,c]
// 256 threads = 8 waves; each wave computes a 32x32 tile (2x2 WMMA 16x16x32).
// Block tile: 64 (M) x 128 (N). Grid: (4096/128, 4096/64).
// Fragment layout (16-bit A/B, wave32): lane holds row/col = lane&15;
// half = lane>>4 selects K-subgroups {half*8..+7} and {16+half*8..+7}.
// ---------------------------------------------------------------------------
__global__ __launch_bounds__(256)
void gemm_kernel(const _Float16* __restrict__ Amat,
                 const _Float16* __restrict__ Bt,
                 const float* __restrict__ W,
                 float* __restrict__ out) {
    const int lane = threadIdx.x & 31;
    const int wave = threadIdx.x >> 5;       // 0..7
    const int half = lane >> 4;              // 0..1
    const int n16  = lane & 15;              // 0..15

    const int i0 = blockIdx.y * 64  + (wave & 1) * 32;   // M tile origin
    const int j0 = blockIdx.x * 128 + (wave >> 1) * 32;  // N tile origin

    const _Float16* aRow0 = Amat + (size_t)(i0 + n16)      * NCOEF;
    const _Float16* aRow1 = Amat + (size_t)(i0 + 16 + n16) * NCOEF;
    const _Float16* bRow0 = Bt   + (size_t)(j0 + n16)      * NCOEF;
    const _Float16* bRow1 = Bt   + (size_t)(j0 + 16 + n16) * NCOEF;
    const int kOff = half * 8;               // this lane's K subgroup start

    v8f acc00 = {}, acc01 = {}, acc10 = {}, acc11 = {};

    for (int kb = 0; kb < NCOEF; kb += 32) {
        const v8h a0lo = *(const v8h*)(aRow0 + kb + kOff);
        const v8h a0hi = *(const v8h*)(aRow0 + kb + 16 + kOff);
        const v8h a1lo = *(const v8h*)(aRow1 + kb + kOff);
        const v8h a1hi = *(const v8h*)(aRow1 + kb + 16 + kOff);
        const v8h b0lo = *(const v8h*)(bRow0 + kb + kOff);
        const v8h b0hi = *(const v8h*)(bRow0 + kb + 16 + kOff);
        const v8h b1lo = *(const v8h*)(bRow1 + kb + kOff);
        const v8h b1hi = *(const v8h*)(bRow1 + kb + 16 + kOff);

        const v16h a0 = __builtin_shufflevector(a0lo, a0hi, 0,1,2,3,4,5,6,7,8,9,10,11,12,13,14,15);
        const v16h a1 = __builtin_shufflevector(a1lo, a1hi, 0,1,2,3,4,5,6,7,8,9,10,11,12,13,14,15);
        const v16h b0 = __builtin_shufflevector(b0lo, b0hi, 0,1,2,3,4,5,6,7,8,9,10,11,12,13,14,15);
        const v16h b1 = __builtin_shufflevector(b1lo, b1hi, 0,1,2,3,4,5,6,7,8,9,10,11,12,13,14,15);

        acc00 = __builtin_amdgcn_wmma_f32_16x16x32_f16(false, a0, false, b0, (short)0, acc00, false, false);
        acc01 = __builtin_amdgcn_wmma_f32_16x16x32_f16(false, a0, false, b1, (short)0, acc01, false, false);
        acc10 = __builtin_amdgcn_wmma_f32_16x16x32_f16(false, a1, false, b0, (short)0, acc10, false, false);
        acc11 = __builtin_amdgcn_wmma_f32_16x16x32_f16(false, a1, false, b1, (short)0, acc11, false, false);
    }

    // Epilogue: C/D layout — lane owns column n16 (+16 for n-tile 1); VGPR v
    // owns row half*8 + v within the 16x16 tile.
    #pragma unroll
    for (int v = 0; v < 8; ++v) {
        const int r0 = i0 + half * 8 + v;        // m tile 0
        const int r1 = r0 + 16;                  // m tile 1
        const int c0 = j0 + n16;                 // n tile 0
        const int c1 = c0 + 16;                  // n tile 1
        size_t o;
        o = (size_t)r0 * NDIM + c0;  out[o] = W[o] + acc00[v];
        o = (size_t)r0 * NDIM + c1;  out[o] = W[o] + acc01[v];
        o = (size_t)r1 * NDIM + c0;  out[o] = W[o] + acc10[v];
        o = (size_t)r1 * NDIM + c1;  out[o] = W[o] + acc11[v];
    }
}

extern "C" void kernel_launch(void* const* d_in, const int* in_sizes, int n_in,
                              void* d_out, int out_size, void* d_ws, size_t ws_size,
                              hipStream_t stream) {
    const float* weight = (const float*)d_in[0];   // [4096, 4096] f32
    const float* dW     = (const float*)d_in[1];   // [768] f32
    const int*   fidx   = (const int*)d_in[2];     // [768] int
    float*       out    = (float*)d_out;           // [4096, 4096] f32

    _Float16* Amat = (_Float16*)d_ws;                       // 4096*768 f16 (6.3 MB)
    _Float16* Bt   = Amat + (size_t)MDIM * NCOEF;           // 4096*768 f16 (6.3 MB)

    prep_kernel<<<(MDIM * NCOEF) / 256, 256, 0, stream>>>(dW, fidx, Amat, Bt);

    dim3 grid(NDIM / 128, MDIM / 64);
    gemm_kernel<<<grid, 256, 0, stream>>>(Amat, Bt, weight, out);
}
